// CrossAttentionWithPrior_68513318306204
// MI455X (gfx1250) — matrix operand order repaired and checked by reading, not compiled, not run on hardware
//
#include <hip/hip_runtime.h>
#include <hip/hip_bf16.h>

typedef _Float16 v16h __attribute__((ext_vector_type(16)));
typedef _Float16 v8h  __attribute__((ext_vector_type(8)));
typedef float    v8f  __attribute__((ext_vector_type(8)));

#define BATCH 8
#define CDIM  512
#define NHEAD 4
#define HDIM  128
#define SQ    2048
#define SQ2   1024
#define SKV   512

__constant__ float c_invtemp[4] = {1.0f, 0.25f, 1.0f/7.0f, 0.1f};

constexpr float PSF    = 0.31622776601683794f;   // sqrt(0.1)
constexpr float QSCALE = 0.08838834764831845f;   // 128^-0.5

// ---------------------------------------------------------------- WMMA helpers
__device__ __forceinline__ v8f wmma16(v16h a, v16h b, v8f c) {
  return __builtin_amdgcn_wmma_f32_16x16x32_f16(false, a, false, b, (short)0, c, false, false);
}

// K order this code uses for A-fragments (16x32), matching its B loads: lane<16 -> {0..15}, lane>=16 -> {16..31}
__device__ __forceinline__ int kpat(int lane, int i) {
  return (lane < 16) ? i : i + 16;
}

// A-fragment gathered from a row-major matrix (row = per-lane M row pointer)
__device__ __forceinline__ v16h frag_a_row(const _Float16* row, int k0, int lane) {
  int sel = (lane >= 16) ? 16 : 0;
  v8h lo = *reinterpret_cast<const v8h*>(row + k0 + sel);
  v8h hi = *reinterpret_cast<const v8h*>(row + k0 + 8 + sel);
  v16h r;
#pragma unroll
  for (int i = 0; i < 8; ++i) { r[i] = lo[i]; r[8 + i] = hi[i]; }
  return r;
}

__device__ __forceinline__ float wave_sum(float v) {
#pragma unroll
  for (int m = 16; m >= 1; m >>= 1) v += __shfl_xor(v, m);
  return v;
}
__device__ __forceinline__ float wave_max(float v) {
#pragma unroll
  for (int m = 16; m >= 1; m >>= 1) v = fmaxf(v, __shfl_xor(v, m));
  return v;
}

// ---------------------------------------------------------------- weight packing
// Packed A layout: frag f = mt*KT + kt ; element ((f*32 + lane)*16 + i) holds
// W[m = mt*16 + lane%16][k = kt*32 + kpat(lane,i)]
__global__ void pack_dense_w(const float* __restrict__ W, _Float16* __restrict__ out) {
  int idx = blockIdx.x * blockDim.x + threadIdx.x;     // 32 mtiles * 16 ktiles * 32 lanes
  if (idx >= 32 * 16 * 32) return;
  int lane = idx & 31, f = idx >> 5;
  int mt = f >> 4, kt = f & 15;
  int m = mt * 16 + (lane & 15);
  size_t base = (size_t)(f * 32 + lane) * 16;
#pragma unroll
  for (int i = 0; i < 16; ++i) {
    int k = kt * 32 + kpat(lane, i);
    out[base + i] = (_Float16)W[(size_t)m * 512 + k];
  }
}

// Conv weights cw[c][ci][3] -> A[m=c][k=dk*512+ci], KT=48
__global__ void pack_conv_w(const float* __restrict__ cw, _Float16* __restrict__ out) {
  int idx = blockIdx.x * blockDim.x + threadIdx.x;     // 32 * 48 * 32
  if (idx >= 32 * 48 * 32) return;
  int lane = idx & 31, f = idx >> 5;
  int mt = f / 48, kt = f % 48;
  int m = mt * 16 + (lane & 15);
  int dk = kt >> 4;
  int cib = (kt & 15) * 32;
  size_t base = (size_t)(f * 32 + lane) * 16;
#pragma unroll
  for (int i = 0; i < 16; ++i) {
    int ci = cib + kpat(lane, i);
    out[base + i] = (_Float16)cw[((size_t)m * 512 + ci) * 3 + dk];
  }
}

// ---------------------------------------------------------------- converts / pads
__global__ void convert_pad_k(const float* __restrict__ in, _Float16* __restrict__ out, int L) {
  int Lp = L + 2;
  size_t total = (size_t)BATCH * Lp * 512;
  for (size_t idx = (size_t)blockIdx.x * blockDim.x + threadIdx.x; idx < total;
       idx += (size_t)gridDim.x * blockDim.x) {
    int c = idx & 511;
    size_t row = idx >> 9;
    int p = (int)(row % Lp);
    int b = (int)(row / Lp);
    float v = (p == 0 || p == Lp - 1) ? 0.0f : in[((size_t)b * L + p - 1) * 512 + c];
    out[idx] = (_Float16)v;
  }
}

__global__ void convert_k(const float* __restrict__ in, _Float16* __restrict__ out, size_t total) {
  for (size_t idx = (size_t)blockIdx.x * blockDim.x + threadIdx.x; idx < total;
       idx += (size_t)gridDim.x * blockDim.x)
    out[idx] = (_Float16)in[idx];
}

__global__ void zero_pad_rows_k(_Float16* __restrict__ p, int Lp) {
  int row = blockIdx.x ? (Lp - 1) : 0;
  p[((size_t)blockIdx.y * Lp + row) * 512 + threadIdx.x] = (_Float16)0.0f;
}

// ---------------------------------------------------------------- conv1d as implicit-GEMM WMMA
// x_h: padded [B][L+2][512] f16 ; wp: packed A (KT=48) ; y: [B][L][512] f32 (+bias)
__global__ void conv1d_wmma_k(const _Float16* __restrict__ x_h, const _Float16* __restrict__ wp,
                              const float* __restrict__ cb, float* __restrict__ y, int L) {
  int lane = threadIdx.x & 31, wid = threadIdx.x >> 5;   // 8 waves
  int nt = blockIdx.x % (L / 16);
  int b  = blockIdx.x / (L / 16);
  int l0 = nt * 16;
  int mt0 = wid * 4;                                     // 4 m-tiles per wave -> 32 total
  const _Float16* xb = x_h + (size_t)b * (L + 2) * 512;
  v8f acc[4] = {};
  int nrow = l0 + (lane & 15);
  int koff = (lane >= 16) ? 16 : 0;
#pragma unroll 4
  for (int kt = 0; kt < 48; ++kt) {
    int dk = kt >> 4;
    int ci0 = (kt & 15) * 32;
    const _Float16* bp = xb + (size_t)(nrow + dk) * 512 + ci0 + koff;
    __builtin_prefetch(bp + 512, 0, 1);
    v16h bf = *reinterpret_cast<const v16h*>(bp);
#pragma unroll
    for (int t = 0; t < 4; ++t) {
      v16h af = *reinterpret_cast<const v16h*>(wp + ((size_t)((mt0 + t) * 48 + kt) * 32 + lane) * 16);
      acc[t] = wmma16(af, bf, acc[t]);
    }
  }
  int msel = (lane >= 16) ? 8 : 0;
#pragma unroll
  for (int t = 0; t < 4; ++t) {
    int cbase = (mt0 + t) * 16 + msel;
    float* orow = y + ((size_t)b * L + nrow) * 512 + cbase;
#pragma unroll
    for (int r = 0; r < 8; ++r) orow[r] = acc[t][r] + cb[cbase + r];
  }
}

// ---------------------------------------------------------------- silu + channel-LN + maxpool(3,2,1)
// y: [B][L][512] f32 -> out_h: padded [B][Lout+2][512] f16 (rows 1..Lout written here)
__global__ void silu_ln_pool_k(const float* __restrict__ y, const float* __restrict__ g,
                               const float* __restrict__ bb, _Float16* __restrict__ out_h,
                               int L, int Lout) {
  int lane = threadIdx.x & 31, wid = threadIdx.x >> 5;   // 4 waves, 1 pooled pos each
  int lo = blockIdx.x * 4 + wid;
  int b = blockIdx.y;
  if (lo >= Lout) return;
  float gv[16], bv[16], rm[16], t[16];
#pragma unroll
  for (int i = 0; i < 16; ++i) {
    int c = lane * 16 + i;
    gv[i] = g[c]; bv[i] = bb[c]; rm[i] = -3.0e38f;
  }
#pragma unroll
  for (int s = 0; s < 3; ++s) {
    int l = 2 * lo - 1 + s;
    if (l < 0 || l >= L) continue;
    const float* row = y + ((size_t)b * L + l) * 512 + lane * 16;
    float sum = 0.f, sq = 0.f;
#pragma unroll
    for (int i = 0; i < 16; ++i) {
      float v = row[i];
      float sv = v / (1.0f + expf(-v));   // silu
      t[i] = sv; sum += sv; sq += sv * sv;
    }
    sum = wave_sum(sum); sq = wave_sum(sq);
    float mean = sum * (1.0f / 512.0f);
    float var = sq * (1.0f / 512.0f) - mean * mean;
    float rs = rsqrtf(var + 1e-5f);
#pragma unroll
    for (int i = 0; i < 16; ++i) {
      float z = (t[i] - mean) * rs * gv[i] + bv[i];
      rm[i] = fmaxf(rm[i], z);
    }
  }
  _Float16* orow = out_h + ((size_t)b * (Lout + 2) + lo + 1) * 512 + lane * 16;
#pragma unroll
  for (int i = 0; i < 16; ++i) orow[i] = (_Float16)rm[i];
}

// qx = x0 + repeat(x1, 2)  (both padded, read rows +1)
__global__ void combine_k(const _Float16* __restrict__ x0, const _Float16* __restrict__ x1,
                          _Float16* __restrict__ qx) {
  size_t total = (size_t)BATCH * SQ2 * 512;
  for (size_t idx = (size_t)blockIdx.x * blockDim.x + threadIdx.x; idx < total;
       idx += (size_t)gridDim.x * blockDim.x) {
    int c = idx & 511;
    size_t row = idx >> 9;
    int i = (int)(row % SQ2);
    int b = (int)(row / SQ2);
    float v0 = (float)x0[((size_t)b * (SQ2 + 2) + i + 1) * 512 + c];
    float v1 = (float)x1[((size_t)b * (SKV + 2) + (i >> 1) + 1) * 512 + c];
    qx[idx] = (_Float16)(v0 + v1);
  }
}

// ---------------------------------------------------------------- dense GEMM Y[n,m]=sum_k X[n,k]W[m,k]
// MODE 0: q (+freq, *scale -> q_h [B,H,SQ2,128])   1: k (+freq -> k_h [B,H,SKV,128])
// MODE 2: v (-> vt_h [B,H,128,SKV] transposed)     3: proj (+bias, dup rows -> xout f32)
template <int MODE>
__global__ void gemm_wmma_k(const _Float16* __restrict__ X, const _Float16* __restrict__ Wp,
                            const float* __restrict__ aux, const float* __restrict__ bias,
                            void* __restrict__ out0) {
  int lane = threadIdx.x & 31, wid = threadIdx.x >> 5;   // 4 waves
  int n0 = blockIdx.x * 16;
  int mt0 = (blockIdx.y * 4 + wid) * 4;                  // 4 m-tiles per wave
  v8f acc[4] = {};
  const _Float16* xrow = X + (size_t)(n0 + (lane & 15)) * 512 + ((lane >= 16) ? 16 : 0);
#pragma unroll 4
  for (int kt = 0; kt < 16; ++kt) {
    v16h bf = *reinterpret_cast<const v16h*>(xrow + kt * 32);
#pragma unroll
    for (int t = 0; t < 4; ++t) {
      v16h af = *reinterpret_cast<const v16h*>(Wp + ((size_t)((mt0 + t) * 16 + kt) * 32 + lane) * 16);
      acc[t] = wmma16(af, bf, acc[t]);
    }
  }
  int n = n0 + (lane & 15);
  int msel = (lane >= 16) ? 8 : 0;
#pragma unroll
  for (int t = 0; t < 4; ++t) {
#pragma unroll
    for (int r = 0; r < 8; ++r) {
      int c = (mt0 + t) * 16 + msel + r;
      float v = acc[t][r];
      if constexpr (MODE == 0) {
        int b = n >> 10, i = n & 1023, h = c >> 7, d = c & 127;
        float qf = aux[((size_t)b * SQ + 2 * i + 1) * 128 + d];
        v = (v + qf * PSF) * QSCALE;
        ((_Float16*)out0)[(((size_t)b * 4 + h) * SQ2 + i) * 128 + d] = (_Float16)v;
      } else if constexpr (MODE == 1) {
        int b = n >> 9, j = n & 511, h = c >> 7, d = c & 127;
        float kf = aux[((size_t)b * SKV + j) * 128 + d];
        v = (v + kf * PSF) * QSCALE;
        ((_Float16*)out0)[(((size_t)b * 4 + h) * SKV + j) * 128 + d] = (_Float16)v;
      } else if constexpr (MODE == 2) {
        int b = n >> 9, j = n & 511, h = c >> 7, d = c & 127;
        ((_Float16*)out0)[(((size_t)b * 4 + h) * 128 + d) * SKV + j] = (_Float16)v;
      } else {
        int b = n >> 10, i = n & 1023;
        v += bias[c];
        float* o = (float*)out0;
        size_t base = ((size_t)b * SQ + 2 * i) * 512 + c;
        o[base] = v; o[base + 512] = v;   // nearest-upsample row duplication
      }
    }
  }
}

// ---------------------------------------------------------------- row squared norms (HD=128)
__global__ void row_norm_k(const _Float16* __restrict__ a, float* __restrict__ out, int rows) {
  int lane = threadIdx.x & 31, wid = threadIdx.x >> 5;
  int row = blockIdx.x * 4 + wid;
  if (row >= rows) return;
  const _Float16* p = a + (size_t)row * 128 + lane * 4;
  float s = 0.f;
#pragma unroll
  for (int i = 0; i < 4; ++i) { float v = (float)p[i]; s += v * v; }
  s = wave_sum(s);
  if (lane == 0) out[row] = s;
}

// ---------------------------------------------------------------- QK^T -> gaussian scores
__global__ void qk_wmma_k(const _Float16* __restrict__ qh, const _Float16* __restrict__ kh,
                          const float* __restrict__ x2, const float* __restrict__ y2,
                          float* __restrict__ attn0) {
  int lane = threadIdx.x & 31;
  int z = blockIdx.z;                  // b*4+h
  int i0 = blockIdx.y * 32;
  int j0 = blockIdx.x * 32;
  const _Float16* qb = qh + (size_t)z * SQ2 * 128;
  const _Float16* kb = kh + (size_t)z * SKV * 128;
  v8f acc[2][2] = {};
  int koff = (lane >= 16) ? 16 : 0;
#pragma unroll
  for (int k0 = 0; k0 < 128; k0 += 32) {
    v16h a0 = frag_a_row(qb + (size_t)(i0 + (lane & 15)) * 128, k0, lane);
    v16h a1 = frag_a_row(qb + (size_t)(i0 + 16 + (lane & 15)) * 128, k0, lane);
    v16h b0 = *reinterpret_cast<const v16h*>(kb + (size_t)(j0 + (lane & 15)) * 128 + k0 + koff);
    v16h b1 = *reinterpret_cast<const v16h*>(kb + (size_t)(j0 + 16 + (lane & 15)) * 128 + k0 + koff);
    acc[0][0] = wmma16(a0, b0, acc[0][0]);
    acc[0][1] = wmma16(a0, b1, acc[0][1]);
    acc[1][0] = wmma16(a1, b0, acc[1][0]);
    acc[1][1] = wmma16(a1, b1, acc[1][1]);
  }
  float it = c_invtemp[z & 3];
  int msel = (lane >= 16) ? 8 : 0;
  const float* x2b = x2 + (size_t)z * SQ2;
  const float* y2b = y2 + (size_t)z * SKV;
#pragma unroll
  for (int a = 0; a < 2; ++a)
#pragma unroll
    for (int bb2 = 0; bb2 < 2; ++bb2) {
      int j = j0 + bb2 * 16 + (lane & 15);
      float yv = y2b[j];
#pragma unroll
      for (int r = 0; r < 8; ++r) {
        int i = i0 + a * 16 + r + msel;
        float d2 = x2b[i] + yv - 2.0f * acc[a][bb2][r];
        attn0[((size_t)z * SQ2 + i) * SKV + j] = -fmaxf(d2, 0.0f) * it;
      }
    }
}

// ---------------------------------------------------------------- 45x5 head-mixing score conv
// logits = attn0 + conv2d(attn0, score_w) + score_b + mask  (tile 16 x 32, all 4 heads)
__global__ void score_conv_k(const float* __restrict__ attn0, const float* __restrict__ sw,
                             const float* __restrict__ sb, const float* __restrict__ mask,
                             float* __restrict__ logits) {
  __shared__ float slab[4][60][36];      // 4 heads x (16+44) x (32+4)
  __shared__ float wl[4 * 45 * 5 * 4];   // [hi][di][dj][h]
  int b = blockIdx.z;
  int i0 = blockIdx.y * 16;
  int j0 = blockIdx.x * 32;
  int tid = threadIdx.x;                 // 256
  for (int idx = tid; idx < 4 * 45 * 5; idx += 256) {
    int hi = idx / 225, rem = idx % 225, di = rem / 5, dj = rem % 5;
#pragma unroll
    for (int h = 0; h < 4; ++h)
      wl[(idx)*4 + h] = sw[(((size_t)h * 4 + hi) * 45 + di) * 5 + dj];
  }
  for (int idx = tid; idx < 4 * 60 * 36; idx += 256) {
    int hi = idx / (60 * 36), rem = idx % (60 * 36), ri = rem / 36, cj = rem % 36;
    int gi = i0 - 22 + ri, gj = j0 - 2 + cj;
    float v = 0.f;
    if (gi >= 0 && gi < SQ2 && gj >= 0 && gj < SKV)
      v = attn0[(((size_t)b * 4 + hi) * SQ2 + gi) * SKV + gj];
    slab[hi][ri][cj] = v;
  }
  __syncthreads();
  int jj = tid & 31;
  int ip = tid >> 5;                     // 0..7 -> 2 rows each
  float acc[8] = {};
  for (int hi = 0; hi < 4; ++hi)
    for (int di = 0; di < 45; ++di) {
#pragma unroll
      for (int dj = 0; dj < 5; ++dj) {
        float s0 = slab[hi][ip * 2 + di][jj + dj];
        float s1 = slab[hi][ip * 2 + 1 + di][jj + dj];
        const float* w4 = &wl[((hi * 45 + di) * 5 + dj) * 4];
#pragma unroll
        for (int h = 0; h < 4; ++h) {
          acc[h * 2 + 0] = fmaf(s0, w4[h], acc[h * 2 + 0]);
          acc[h * 2 + 1] = fmaf(s1, w4[h], acc[h * 2 + 1]);
        }
      }
    }
#pragma unroll
  for (int h = 0; h < 4; ++h)
#pragma unroll
    for (int p = 0; p < 2; ++p) {
      int i = i0 + ip * 2 + p, j = j0 + jj;
      float center = slab[h][ip * 2 + p + 22][jj + 2];
      float v = center + acc[h * 2 + p] + sb[h] + mask[((size_t)b * SQ2 + i) * SKV + j];
      logits[(((size_t)b * 4 + h) * SQ2 + i) * SKV + j] = v;
    }
}

// ---------------------------------------------------------------- softmax + prior fusion + renorm
__global__ void softmax_prior_k(const float* __restrict__ logits, const float* __restrict__ prior,
                                float* __restrict__ attn_out, _Float16* __restrict__ attn_h) {
  int lane = threadIdx.x & 31;
  int bhi = blockIdx.x;                  // B*H*SQ2 rows
  int i = bhi & 1023;
  int h = (bhi >> 10) & 3;
  int b = bhi >> 12;
  const float* row = logits + (size_t)bhi * SKV + lane * 16;
  float l[16];
  float m = -3.0e38f;
#pragma unroll
  for (int t = 0; t < 16; ++t) { l[t] = row[t]; m = fmaxf(m, l[t]); }
  m = wave_max(m);
  float s = 0.f;
#pragma unroll
  for (int t = 0; t < 16; ++t) { l[t] = expf(l[t] - m); s += l[t]; }
  s = wave_sum(s);
  float inv = 1.0f / s;
  const float* pr = prior + ((size_t)b * SQ + 2 * i + 1) * SKV + lane * 16;
  float s2 = 0.f;
#pragma unroll
  for (int t = 0; t < 16; ++t) {
    float p = fmaxf(l[t] * inv, 1e-8f) * fmaxf(pr[t], 1e-8f);
    l[t] = p; s2 += p;
  }
  s2 = wave_sum(s2);
  float inv2 = 1.0f / (s2 + 1e-8f);
  size_t obase = (((size_t)b * 4 + h) * SQ + 2 * i) * SKV + lane * 16;
  _Float16* hrow = attn_h + (size_t)bhi * SKV + lane * 16;
#pragma unroll
  for (int t = 0; t < 16; ++t) {
    float o = l[t] * inv2;
    attn_out[obase + t] = o;             // row 2i
    attn_out[obase + SKV + t] = o;       // row 2i+1 (nearest upsample)
    hrow[t] = (_Float16)o;
  }
}

// ---------------------------------------------------------------- attn @ V (K=512)
__global__ void av_wmma_k(const _Float16* __restrict__ ah, const _Float16* __restrict__ vt,
                          _Float16* __restrict__ ctx) {
  int lane = threadIdx.x & 31;
  int z = blockIdx.z;                    // b*4+h
  int i0 = blockIdx.y * 32;
  int d0 = blockIdx.x * 32;
  const _Float16* ab = ah + (size_t)z * SQ2 * SKV;
  const _Float16* vb = vt + (size_t)z * 128 * SKV;
  v8f acc[2][2] = {};
  int koff = (lane >= 16) ? 16 : 0;
#pragma unroll 2
  for (int k0 = 0; k0 < SKV; k0 += 32) {
    v16h a0 = frag_a_row(ab + (size_t)(i0 + (lane & 15)) * SKV, k0, lane);
    v16h a1 = frag_a_row(ab + (size_t)(i0 + 16 + (lane & 15)) * SKV, k0, lane);
    v16h b0 = *reinterpret_cast<const v16h*>(vb + (size_t)(d0 + (lane & 15)) * SKV + k0 + koff);
    v16h b1 = *reinterpret_cast<const v16h*>(vb + (size_t)(d0 + 16 + (lane & 15)) * SKV + k0 + koff);
    acc[0][0] = wmma16(a0, b0, acc[0][0]);
    acc[0][1] = wmma16(a0, b1, acc[0][1]);
    acc[1][0] = wmma16(a1, b0, acc[1][0]);
    acc[1][1] = wmma16(a1, b1, acc[1][1]);
  }
  int msel = (lane >= 16) ? 8 : 0;
  int b = z >> 2, h = z & 3;
#pragma unroll
  for (int a = 0; a < 2; ++a)
#pragma unroll
    for (int dt = 0; dt < 2; ++dt) {
      int d = d0 + dt * 16 + (lane & 15);
#pragma unroll
      for (int r = 0; r < 8; ++r) {
        int i = i0 + a * 16 + r + msel;
        ctx[((size_t)b * SQ2 + i) * 512 + h * 128 + d] = (_Float16)acc[a][dt][r];
      }
    }
}

// ---------------------------------------------------------------- launch
extern "C" void kernel_launch(void* const* d_in, const int* in_sizes, int n_in,
                              void* d_out, int out_size, void* d_ws, size_t ws_size,
                              hipStream_t stream) {
  (void)in_sizes; (void)n_in; (void)out_size; (void)ws_size;
  const float* q_x      = (const float*)d_in[0];
  const float* kv_x     = (const float*)d_in[1];
  const float* q_freqs  = (const float*)d_in[2];
  const float* k_freqs  = (const float*)d_in[3];
  const float* mask     = (const float*)d_in[4];
  const float* a_prior  = (const float*)d_in[5];
  const float* qds_cw   = (const float*)d_in[6];
  const float* qds_cb   = (const float*)d_in[7];
  const float* qds_lg   = (const float*)d_in[8];
  const float* qds_lb   = (const float*)d_in[9];
  const float* qp_cw    = (const float*)d_in[10];   // [2,512,512,3] -> layer 0
  const float* qp_cb    = (const float*)d_in[11];   // [2,512]
  const float* qp_lg    = (const float*)d_in[12];
  const float* qp_lb    = (const float*)d_in[13];
  const float* wq       = (const float*)d_in[14];
  const float* wk       = (const float*)d_in[15];
  const float* wv       = (const float*)d_in[16];
  const float* score_w  = (const float*)d_in[17];
  const float* score_b  = (const float*)d_in[18];
  const float* proj_w   = (const float*)d_in[19];
  const float* proj_b   = (const float*)d_in[20];

  float* xout  = (float*)d_out;                              // [B,SQ,512]
  float* aout  = (float*)d_out + (size_t)BATCH * SQ * 512;   // [B,H,SQ,SKV]

  char* w = (char*)d_ws;
  size_t off = 0;
  auto alloc = [&](size_t bytes) { void* p = w + off; off += (bytes + 255) & ~(size_t)255; return p; };

  _Float16* xq_h  = (_Float16*)alloc((size_t)BATCH * (SQ + 2) * 512 * 2);
  float*    ybuf  = (float*)   alloc((size_t)BATCH * SQ * 512 * 4);       // conv out, reused
  _Float16* x0_h  = (_Float16*)alloc((size_t)BATCH * (SQ2 + 2) * 512 * 2);
  _Float16* x1_h  = (_Float16*)alloc((size_t)BATCH * (SKV + 2) * 512 * 2);
  _Float16* qx_h  = (_Float16*)alloc((size_t)BATCH * SQ2 * 512 * 2);
  _Float16* kv_h  = (_Float16*)alloc((size_t)BATCH * SKV * 512 * 2);
  _Float16* q_h   = (_Float16*)alloc((size_t)BATCH * NHEAD * SQ2 * 128 * 2);
  _Float16* k_h   = (_Float16*)alloc((size_t)BATCH * NHEAD * SKV * 128 * 2);
  _Float16* vt_h  = (_Float16*)alloc((size_t)BATCH * NHEAD * 128 * SKV * 2);
  float*    x2    = (float*)   alloc((size_t)BATCH * NHEAD * SQ2 * 4);
  float*    y2    = (float*)   alloc((size_t)BATCH * NHEAD * SKV * 4);
  float*    attn0 = (float*)   alloc((size_t)BATCH * NHEAD * SQ2 * SKV * 4);
  float*    logit = (float*)   alloc((size_t)BATCH * NHEAD * SQ2 * SKV * 4);
  _Float16* ath   = (_Float16*)alloc((size_t)BATCH * NHEAD * SQ2 * SKV * 2);
  _Float16* ctx_h = (_Float16*)alloc((size_t)BATCH * SQ2 * 512 * 2);
  _Float16* wq_p  = (_Float16*)alloc((size_t)512 * 512 * 2);
  _Float16* wk_p  = (_Float16*)alloc((size_t)512 * 512 * 2);
  _Float16* wv_p  = (_Float16*)alloc((size_t)512 * 512 * 2);
  _Float16* wo_p  = (_Float16*)alloc((size_t)512 * 512 * 2);
  _Float16* cw1_p = (_Float16*)alloc((size_t)512 * 1536 * 2);
  _Float16* cw2_p = (_Float16*)alloc((size_t)512 * 1536 * 2);

  // --- weight packing (A-fragment swizzle)
  pack_dense_w<<<64, 256, 0, stream>>>(wq, wq_p);
  pack_dense_w<<<64, 256, 0, stream>>>(wk, wk_p);
  pack_dense_w<<<64, 256, 0, stream>>>(wv, wv_p);
  pack_dense_w<<<64, 256, 0, stream>>>(proj_w, wo_p);
  pack_conv_w<<<192, 256, 0, stream>>>(qds_cw, cw1_p);
  pack_conv_w<<<192, 256, 0, stream>>>(qp_cw, cw2_p);        // layer 0 slice

  // --- input conversion
  convert_pad_k<<<4096, 256, 0, stream>>>(q_x, xq_h, SQ);
  convert_k<<<4096, 256, 0, stream>>>(kv_x, kv_h, (size_t)BATCH * SKV * 512);

  // --- ds-block 1: conv(WMMA) -> silu+LN+pool
  conv1d_wmma_k<<<BATCH * SQ / 16, 256, 0, stream>>>(xq_h, cw1_p, qds_cb, ybuf, SQ);
  zero_pad_rows_k<<<dim3(2, BATCH), 512, 0, stream>>>(x0_h, SQ2 + 2);
  silu_ln_pool_k<<<dim3(SQ2 / 4, BATCH), 128, 0, stream>>>(ybuf, qds_lg, qds_lb, x0_h, SQ, SQ2);

  // --- ds-block 2 (QueryProj layer 0; layer-1 output is dead code in the reference)
  conv1d_wmma_k<<<BATCH * SQ2 / 16, 256, 0, stream>>>(x0_h, cw2_p, qp_cb, ybuf, SQ2);
  zero_pad_rows_k<<<dim3(2, BATCH), 512, 0, stream>>>(x1_h, SKV + 2);
  silu_ln_pool_k<<<dim3(SKV / 4, BATCH), 128, 0, stream>>>(ybuf, qp_lg, qp_lb, x1_h, SQ2, SKV);

  // --- combine (x0 + 2x nearest-upsampled x1)
  combine_k<<<8192, 256, 0, stream>>>(x0_h, x1_h, qx_h);

  // --- QKV projections (WMMA) with freq/scale epilogues
  gemm_wmma_k<0><<<dim3(BATCH * SQ2 / 16, 2), 128, 0, stream>>>(qx_h, wq_p, q_freqs, nullptr, q_h);
  gemm_wmma_k<1><<<dim3(BATCH * SKV / 16, 2), 128, 0, stream>>>(kv_h, wk_p, k_freqs, nullptr, k_h);
  gemm_wmma_k<2><<<dim3(BATCH * SKV / 16, 2), 128, 0, stream>>>(kv_h, wv_p, nullptr, nullptr, vt_h);

  // --- squared row norms
  row_norm_k<<<BATCH * NHEAD * SQ2 / 4, 128, 0, stream>>>(q_h, x2, BATCH * NHEAD * SQ2);
  row_norm_k<<<BATCH * NHEAD * SKV / 4, 128, 0, stream>>>(k_h, y2, BATCH * NHEAD * SKV);

  // --- gaussian scores via QK^T (WMMA)
  qk_wmma_k<<<dim3(SKV / 32, SQ2 / 32, BATCH * NHEAD), 32, 0, stream>>>(q_h, k_h, x2, y2, attn0);

  // --- 45x5 head-mixing score conv + mask
  score_conv_k<<<dim3(SKV / 32, SQ2 / 16, BATCH), 256, 0, stream>>>(attn0, score_w, score_b, mask, logit);

  // --- softmax + prior fusion + renorm; writes final attn (duplicated rows) + f16 copy
  softmax_prior_k<<<BATCH * NHEAD * SQ2, 32, 0, stream>>>(logit, a_prior, aout, ath);

  // --- attn @ V (WMMA) -> ctx, then output projection (WMMA) with row duplication
  av_wmma_k<<<dim3(128 / 32, SQ2 / 32, BATCH * NHEAD), 32, 0, stream>>>(ath, vt_h, ctx_h);
  gemm_wmma_k<3><<<dim3(BATCH * SQ2 / 16, 2), 128, 0, stream>>>(ctx_h, wo_p, nullptr, proj_b, xout);
}
